// Group_query_Attention_20495583936570
// MI455X (gfx1250) — compile-verified, matrix operand order checked by reading, not run on hardware
//
#include <hip/hip_runtime.h>
#include <stdint.h>

// ---- problem constants (from reference) ----
#define EMBED 2048
#define NQH   32
#define NKVH  8
#define HD    64
#define TSEQ  2048
#define BATCH 2
#define QSCALE 0.125f   // 1/sqrt(64)

typedef __attribute__((ext_vector_type(16))) __bf16 v16bf;
typedef __attribute__((ext_vector_type(8)))  float  v8f;
typedef unsigned short u16;

union FragBF {
  v16bf v;
  uint4 q[2];
};

__device__ __forceinline__ u16 f2bf(float f) {
  unsigned u = __float_as_uint(f);
  u += 0x7FFFu + ((u >> 16) & 1u);     // round-to-nearest-even
  return (u16)(u >> 16);
}

__device__ __forceinline__ v8f wmma_bf16(const FragBF& a, const FragBF& b, v8f c) {
  return __builtin_amdgcn_wmma_f32_16x16x32_bf16(false, a.v, false, b.v, (short)0, c, false, false);
}

__device__ __forceinline__ float rmax16(float v) {
  v = fmaxf(v, __shfl_xor(v, 1, 32));
  v = fmaxf(v, __shfl_xor(v, 2, 32));
  v = fmaxf(v, __shfl_xor(v, 4, 32));
  v = fmaxf(v, __shfl_xor(v, 8, 32));
  return v;
}
__device__ __forceinline__ float rsum16(float v) {
  v += __shfl_xor(v, 1, 32);
  v += __shfl_xor(v, 2, 32);
  v += __shfl_xor(v, 4, 32);
  v += __shfl_xor(v, 8, 32);
  return v;
}

// ---------------- fp32 -> bf16 convert ----------------
__global__ void cvt_f32_bf16(const float* __restrict__ in, u16* __restrict__ out, int n) {
  int i = blockIdx.x * blockDim.x + threadIdx.x;
  int stride = gridDim.x * blockDim.x;
  for (; i < n; i += stride) out[i] = f2bf(in[i]);
}

// ---------------- WMMA GEMM: C = A (MxK) * W^T (W is NxK), bf16 in, f32 acc ----------------
// Per-wave tile: 32(M) x 64(N)  -> 8 WMMAs per 12 b128 loads per K-step.
// MODE 0: store bf16 row-major [M,N] scaled by QSCALE          (Q projection)
// MODE 1: store bf16 into K layout [B,NKVH,T,HD]               (K projection)
// MODE 2: store bf16 TRANSPOSED into V layout [B,NKVH,HD,T]    (V projection)
// MODE 3: store fp32 row-major [M,N]                           (output projection)
template <int MODE>
__global__ __launch_bounds__(128)
void gemm_wmma(const u16* __restrict__ A, const u16* __restrict__ W,
               void* __restrict__ Cout, int N, int K) {
  const int wave = threadIdx.x >> 5;
  const int lane = threadIdx.x & 31;
  const int hf   = lane >> 4;
  const int lm   = lane & 15;
  const int m0 = blockIdx.y * 128 + wave * 32;
  const int n0 = blockIdx.x * 64;

  const v8f vzero = {0.f,0.f,0.f,0.f,0.f,0.f,0.f,0.f};
  v8f acc[2][4];
  #pragma unroll
  for (int mi = 0; mi < 2; ++mi)
    #pragma unroll
    for (int j = 0; j < 4; ++j) acc[mi][j] = vzero;

  const u16* arow0 = A + (size_t)(m0 + lm) * K + 8 * hf;
  const u16* arow1 = A + (size_t)(m0 + 16 + lm) * K + 8 * hf;
  for (int kk = 0; kk < K; kk += 32) {
    FragBF a0, a1;
    const uint4* pa0 = (const uint4*)(arow0 + kk);
    a0.q[0] = pa0[0];          // K = kk+8*hf+0..7
    a0.q[1] = pa0[2];          // K = kk+8*hf+16..23
    const uint4* pa1 = (const uint4*)(arow1 + kk);
    a1.q[0] = pa1[0];
    a1.q[1] = pa1[2];
    #pragma unroll
    for (int j = 0; j < 4; ++j) {
      FragBF b;
      const uint4* pb = (const uint4*)(W + (size_t)(n0 + j * 16 + lm) * K + kk + 8 * hf);
      b.q[0] = pb[0];
      b.q[1] = pb[2];
      acc[0][j] = wmma_bf16(a0, b, acc[0][j]);
      acc[1][j] = wmma_bf16(a1, b, acc[1][j]);
    }
  }

  // C/D layout: lane -> col (lm), vgpr r -> row (mi*16 + r + 8*hf)
  if (MODE == 0) {
    u16* C = (u16*)Cout;
    #pragma unroll
    for (int mi = 0; mi < 2; ++mi)
      #pragma unroll
      for (int j = 0; j < 4; ++j) {
        const int col = n0 + j * 16 + lm;
        #pragma unroll
        for (int r = 0; r < 8; ++r) {
          const int row = m0 + mi * 16 + 8 * hf + r;
          C[(size_t)row * N + col] = f2bf(acc[mi][j][r] * QSCALE);
        }
      }
  } else if (MODE == 1) {
    u16* C = (u16*)Cout;
    #pragma unroll
    for (int mi = 0; mi < 2; ++mi)
      #pragma unroll
      for (int j = 0; j < 4; ++j) {
        const int col = n0 + j * 16 + lm;
        const int kvh = col >> 6, d = col & 63;
        #pragma unroll
        for (int r = 0; r < 8; ++r) {
          const int row = m0 + mi * 16 + 8 * hf + r;
          const int bb = row >> 11, t = row & (TSEQ - 1);
          C[(((size_t)(bb * NKVH + kvh)) * TSEQ + t) * HD + d] = f2bf(acc[mi][j][r]);
        }
      }
  } else if (MODE == 2) {
    u16* C = (u16*)Cout;
    #pragma unroll
    for (int mi = 0; mi < 2; ++mi) {
      const int rowbase = m0 + mi * 16 + 8 * hf;     // 8 consecutive rows (t contiguous)
      const int bb = rowbase >> 11, t0 = rowbase & (TSEQ - 1);
      #pragma unroll
      for (int j = 0; j < 4; ++j) {
        const int col = n0 + j * 16 + lm;
        const int kvh = col >> 6, d = col & 63;
        uint4 pack;
        pack.x = (unsigned)f2bf(acc[mi][j][0]) | ((unsigned)f2bf(acc[mi][j][1]) << 16);
        pack.y = (unsigned)f2bf(acc[mi][j][2]) | ((unsigned)f2bf(acc[mi][j][3]) << 16);
        pack.z = (unsigned)f2bf(acc[mi][j][4]) | ((unsigned)f2bf(acc[mi][j][5]) << 16);
        pack.w = (unsigned)f2bf(acc[mi][j][6]) | ((unsigned)f2bf(acc[mi][j][7]) << 16);
        *(uint4*)(C + ((((size_t)(bb * NKVH + kvh)) * HD + d) * TSEQ + t0)) = pack;
      }
    }
  } else { // MODE 3
    float* C = (float*)Cout;
    #pragma unroll
    for (int mi = 0; mi < 2; ++mi)
      #pragma unroll
      for (int j = 0; j < 4; ++j) {
        const int col = n0 + j * 16 + lm;
        #pragma unroll
        for (int r = 0; r < 8; ++r) {
          const int row = m0 + mi * 16 + 8 * hf + r;
          C[(size_t)row * N + col] = acc[mi][j][r];
        }
      }
  }
}

// ---------------- flash attention, one wave per (b, h, 16 q-rows) ----------------
// Q: [B*T, EMBED] bf16 (scale pre-folded) ; K: [B,NKVH,T,HD] bf16 ; V: [B,NKVH,HD,T] bf16
// O: [B*T, EMBED] bf16
__global__ __launch_bounds__(32)
void attn_kernel(const u16* __restrict__ Q, const u16* __restrict__ Kp,
                 const u16* __restrict__ Vt, u16* __restrict__ O) {
  __shared__ u16 lp[16 * 32];                  // P tile staging (16 q rows x 32 kv cols)
  const int lane = threadIdx.x;
  const int hf = lane >> 4;
  const int lm = lane & 15;
  const int q0 = blockIdx.x * 16;
  const int h  = blockIdx.y;
  const int b  = blockIdx.z;
  const int kvh = h >> 2;                      // GROUP = 4

  // Q A-fragments for the two 32-wide K(=d) chunks
  FragBF aQ[2];
  const u16* qrow = Q + (size_t)(b * TSEQ + q0 + lm) * EMBED + h * HD;
  #pragma unroll
  for (int dc = 0; dc < 2; ++dc) {
    const uint4* p = (const uint4*)(qrow + dc * 32 + 8 * hf);
    aQ[dc].q[0] = p[0];
    aQ[dc].q[1] = p[2];
  }

  const v8f vzero = {0.f,0.f,0.f,0.f,0.f,0.f,0.f,0.f};
  v8f o[4];
  #pragma unroll
  for (int dt = 0; dt < 4; ++dt) o[dt] = vzero;
  float mrow[8], lrow[8];
  #pragma unroll
  for (int r = 0; r < 8; ++r) { mrow[r] = -3.0e30f; lrow[r] = 0.f; }

  const u16* kbase = Kp + (size_t)(b * NKVH + kvh) * TSEQ * HD;
  const u16* vbase = Vt + (size_t)(b * NKVH + kvh) * HD * TSEQ;

  for (int k0 = 0; k0 < q0 + 16; k0 += 32) {
    // ---- S = Q * K^T (two 16-col tiles) ----
    v8f s0 = vzero, s1 = vzero;
    #pragma unroll
    for (int dc = 0; dc < 2; ++dc) {
      FragBF kb;
      const uint4* p0 = (const uint4*)(kbase + (size_t)(k0 + lm) * HD + dc * 32 + 8 * hf);
      kb.q[0] = p0[0]; kb.q[1] = p0[2];
      s0 = wmma_bf16(aQ[dc], kb, s0);
      const uint4* p1 = (const uint4*)(kbase + (size_t)(k0 + 16 + lm) * HD + dc * 32 + 8 * hf);
      kb.q[0] = p1[0]; kb.q[1] = p1[2];
      s1 = wmma_bf16(aQ[dc], kb, s1);
    }

    // ---- causal mask: only the trailing chunk of each q-tile is partial.
    // k0, q0 are wave-uniform -> this is a scalar branch, skipped in the hot loop.
    if (k0 + 31 > q0) {
      const int c0 = k0 + lm;
      const int c1 = k0 + 16 + lm;
      #pragma unroll
      for (int r = 0; r < 8; ++r) {
        const int row = q0 + 8 * hf + r;
        if (c0 > row) s0[r] = -3.0e30f;
        if (c1 > row) s1[r] = -3.0e30f;
      }
    }

    // ---- online softmax (acc layout: lane->col, vgpr r -> row r+8*hf) ----
    #pragma unroll
    for (int r = 0; r < 8; ++r) {
      const float tmax = rmax16(fmaxf(s0[r], s1[r]));
      const float mn = fmaxf(mrow[r], tmax);
      const float corr = __expf(mrow[r] - mn);
      const float p0 = __expf(s0[r] - mn);
      const float p1 = __expf(s1[r] - mn);
      lrow[r] = lrow[r] * corr + rsum16(p0 + p1);
      mrow[r] = mn;
      #pragma unroll
      for (int dt = 0; dt < 4; ++dt) o[dt][r] *= corr;   // corr uniform across the 16-lane group
      lp[(8 * hf + r) * 32 + lm]      = f2bf(p0);
      lp[(8 * hf + r) * 32 + 16 + lm] = f2bf(p1);
    }
    asm volatile("s_wait_dscnt 0" ::: "memory");

    // ---- reload P as A-fragment (transpose via LDS), O += P * V ----
    FragBF aP;
    const uint4* pp = (const uint4*)(lp + lm * 32 + 8 * hf);
    aP.q[0] = pp[0]; aP.q[1] = pp[2];
    #pragma unroll
    for (int dt = 0; dt < 4; ++dt) {
      FragBF vb;
      const uint4* pv = (const uint4*)(vbase + (size_t)(dt * 16 + lm) * TSEQ + k0 + 8 * hf);
      vb.q[0] = pv[0]; vb.q[1] = pv[2];
      o[dt] = wmma_bf16(aP, vb, o[dt]);
    }
  }

  // ---- normalize and store O (bf16) ----
  #pragma unroll
  for (int r = 0; r < 8; ++r) {
    const float inv = 1.0f / lrow[r];
    const size_t rowoff = (size_t)(b * TSEQ + q0 + 8 * hf + r) * EMBED + h * HD;
    #pragma unroll
    for (int dt = 0; dt < 4; ++dt)
      O[rowoff + dt * 16 + lm] = f2bf(o[dt][r] * inv);
  }
}

// ---------------- host launcher ----------------
extern "C" void kernel_launch(void* const* d_in, const int* in_sizes, int n_in,
                              void* d_out, int out_size, void* d_ws, size_t ws_size,
                              hipStream_t stream) {
  const float* x  = (const float*)d_in[0];
  const float* wq = (const float*)d_in[1];
  const float* wk = (const float*)d_in[2];
  const float* wv = (const float*)d_in[3];
  const float* wo = (const float*)d_in[4];

  const int M = BATCH * TSEQ;                 // 4096 rows
  char* ws = (char*)d_ws;
  size_t off = 0;
  auto alloc = [&](size_t bytes) -> void* {
    void* p = ws + off;
    off += (bytes + 255) & ~(size_t)255;
    return p;
  };
  u16* xb  = (u16*)alloc((size_t)M * EMBED * 2);
  u16* wqb = (u16*)alloc((size_t)EMBED * EMBED * 2);
  u16* wkb = (u16*)alloc((size_t)NKVH * HD * EMBED * 2);
  u16* wvb = (u16*)alloc((size_t)NKVH * HD * EMBED * 2);
  u16* wob = (u16*)alloc((size_t)EMBED * EMBED * 2);
  u16* Qb  = (u16*)alloc((size_t)M * EMBED * 2);
  u16* Kb  = (u16*)alloc((size_t)BATCH * NKVH * TSEQ * HD * 2);
  u16* Vtb = (u16*)alloc((size_t)BATCH * NKVH * HD * TSEQ * 2);
  u16* Ob  = xb;  // x no longer needed after the three projection GEMMs -> reuse for attention output

  const int CB = 256;
  cvt_f32_bf16<<<dim3((M * EMBED + CB - 1) / CB), CB, 0, stream>>>(x,  xb,  M * EMBED);
  cvt_f32_bf16<<<dim3((EMBED * EMBED + CB - 1) / CB), CB, 0, stream>>>(wq, wqb, EMBED * EMBED);
  cvt_f32_bf16<<<dim3((NKVH * HD * EMBED + CB - 1) / CB), CB, 0, stream>>>(wk, wkb, NKVH * HD * EMBED);
  cvt_f32_bf16<<<dim3((NKVH * HD * EMBED + CB - 1) / CB), CB, 0, stream>>>(wv, wvb, NKVH * HD * EMBED);
  cvt_f32_bf16<<<dim3((EMBED * EMBED + CB - 1) / CB), CB, 0, stream>>>(wo, wob, EMBED * EMBED);

  // Q = xb * wq^T (scaled), K -> [B,NKVH,T,HD], V -> [B,NKVH,HD,T]
  gemm_wmma<0><<<dim3(EMBED / 64, M / 128), 128, 0, stream>>>(xb, wqb, Qb, EMBED, EMBED);
  gemm_wmma<1><<<dim3((NKVH * HD) / 64, M / 128), 128, 0, stream>>>(xb, wkb, Kb, NKVH * HD, EMBED);
  gemm_wmma<2><<<dim3((NKVH * HD) / 64, M / 128), 128, 0, stream>>>(xb, wvb, Vtb, NKVH * HD, EMBED);

  // flash attention (overwrites Ob == xb region)
  attn_kernel<<<dim3(TSEQ / 16, NQH, BATCH), 32, 0, stream>>>(Qb, Kb, Vtb, Ob);

  // final projection to fp32 output
  gemm_wmma<3><<<dim3(EMBED / 64, M / 128), 128, 0, stream>>>(Ob, wob, d_out, EMBED, EMBED);

  (void)in_sizes; (void)n_in; (void)out_size; (void)ws_size;
}
